// AgentLayer_55843164783332
// MI455X (gfx1250) — matrix operand order repaired and checked by reading, not compiled, not run on hardware
//
#include <hip/hip_runtime.h>
#include <hip/hip_bf16.h>

#define B_   512
#define T_   128
#define D_   256
#define S_   64
#define H_   256
#define K_   10
#define U_   64
#define G3H  768
#define LAMBDA 0.5f

typedef __attribute__((ext_vector_type(16))) __bf16 v16bf;
typedef __attribute__((ext_vector_type(8)))  __bf16 v8bf;
typedef __attribute__((ext_vector_type(8)))  float  v8f;

// ---------------- helpers ----------------

__device__ __forceinline__ unsigned short f2bf_bits(float f) {
    unsigned u = __builtin_bit_cast(unsigned, f);
    unsigned r = (u + 0x7FFFu + ((u >> 16) & 1u)) >> 16;   // round-to-nearest-even
    return (unsigned short)r;
}

__device__ __forceinline__ unsigned rotl32(unsigned x, int r) {
    return (x << r) | (x >> (32 - r));
}

// threefry2x32-style 20-round mixer (stateless, deterministic)
__device__ __forceinline__ unsigned threefry(unsigned k0, unsigned k1,
                                             unsigned c0, unsigned c1) {
    const unsigned ks2 = 0x1BD11BDAu ^ k0 ^ k1;
    unsigned x0 = c0 + k0, x1 = c1 + k1;
    const int R0[4] = {13, 15, 26, 6};
    const int R1[4] = {17, 29, 16, 24};
#pragma unroll
    for (int j = 0; j < 4; j++) { x0 += x1; x1 = rotl32(x1, R0[j]); x1 ^= x0; }
    x0 += k1;  x1 += ks2 + 1u;
#pragma unroll
    for (int j = 0; j < 4; j++) { x0 += x1; x1 = rotl32(x1, R1[j]); x1 ^= x0; }
    x0 += ks2; x1 += k0 + 2u;
#pragma unroll
    for (int j = 0; j < 4; j++) { x0 += x1; x1 = rotl32(x1, R0[j]); x1 ^= x0; }
    x0 += k0;  x1 += k1 + 3u;
#pragma unroll
    for (int j = 0; j < 4; j++) { x0 += x1; x1 = rotl32(x1, R1[j]); x1 ^= x0; }
    x0 += k1;  x1 += ks2 + 4u;
#pragma unroll
    for (int j = 0; j < 4; j++) { x0 += x1; x1 = rotl32(x1, R0[j]); x1 ^= x0; }
    x0 += ks2; x1 += k0 + 5u;
    return x0 ^ x1;
}

// A fragment from a bf16 row (16-bit A, 16x32 tile):
// lanes 0-15 row=lane, halves K {0..7},{16..23}; lanes 16-31 K {8..15},{24..31}.
// Pure loads: two 16-byte global_load_b128, no conversion VALU.
__device__ __forceinline__ v16bf load_a_bf(const unsigned short* __restrict__ ar,
                                           int k0, int half) {
    const v8bf* p0 = reinterpret_cast<const v8bf*>(ar + k0 + 8 * half);
    const v8bf* p1 = reinterpret_cast<const v8bf*>(ar + k0 + 16 + 8 * half);
    v8bf lo = *p0, hi = *p1;
    return __builtin_shufflevector(lo, hi, 0, 1, 2, 3, 4, 5, 6, 7,
                                           8, 9, 10, 11, 12, 13, 14, 15);
}

// B fragment from pre-packed buffer: contiguous 16 bf16 per lane (32B load)
__device__ __forceinline__ v16bf load_b_frag(const unsigned short* __restrict__ packed,
                                             int Kdim, int tn, int kc, int lane) {
    const v16bf* p = reinterpret_cast<const v16bf*>(
        packed + (((size_t)tn * (Kdim >> 5) + kc) << 9) + ((size_t)lane << 4));
    return *p;
}

// ---------------- elementwise fp32 -> bf16 mirror ----------------
__global__ void cvt_bf_kernel(const float* __restrict__ src,
                              unsigned short* __restrict__ dst, int n) {
    int idx = blockIdx.x * blockDim.x + threadIdx.x;
    if (idx < n) dst[idx] = f2bf_bits(src[idx]);
}

// ---------------- weight packing ----------------
// Packs W^T (B-matrix, [K][N], B[k][n] = w[n*K + k]) into WMMA B-fragment order:
// packed[(tn*(K/32)+kc)*512 + lane*16 + i] = B[kc*32 + (i<8 ? 8*(lane>>4) + i : 16 + 8*(lane>>4) + i-8)][tn*16 + (lane&15)]
__global__ void pack_b_kernel(const float* __restrict__ w, unsigned short* __restrict__ dst,
                              int N, int Kdim) {
    int p = blockIdx.x * blockDim.x + threadIdx.x;
    if (p >= N * Kdim) return;
    int i    = p & 15;
    int lane = (p >> 4) & 31;
    int rest = p >> 9;
    int kcN  = Kdim >> 5;
    int kc   = rest % kcN;
    int tn   = rest / kcN;
    int k = kc * 32 + ((i < 8) ? (((lane >> 4) << 3) + i) : (16 + ((lane >> 4) << 3) + i - 8));
    int n = tn * 16 + (lane & 15);
    dst[p] = f2bf_bits(w[(size_t)n * Kdim + k]);
}

// ---------------- GX = X @ w_ih^T + b_ih  (all timesteps, WMMA) ----------------
// GX layout [T][B][3H], row m = t*B + b ; A row from xbf[b][t][:]
__global__ void gemm_gx_kernel(const unsigned short* __restrict__ xbf,
                               const unsigned short* __restrict__ pW,
                               const float* __restrict__ bias,
                               float* __restrict__ gx) {
    int lane = threadIdx.x & 31;
    int wid  = threadIdx.x >> 5;
    int wg   = blockIdx.x * 8 + wid;          // 4096 Mtiles * 12 Ngroups = 49152
    int ng   = wg % 12;
    int mt   = wg / 12;
    int half = lane >> 4, row = lane & 15;
    int mrow = mt * 16 + row;
    int b = mrow & (B_ - 1);
    int t = mrow >> 9;
    const unsigned short* ar = xbf + ((size_t)b * T_ + t) * D_;
    v8f acc[4] = {};
#pragma unroll
    for (int kc = 0; kc < 8; ++kc) {
        v16bf a = load_a_bf(ar, kc * 32, half);
#pragma unroll
        for (int j = 0; j < 4; j++) {
            v16bf bf = load_b_frag(pW, 256, ng * 4 + j, kc, lane);
            acc[j] = __builtin_amdgcn_wmma_f32_16x16x32_bf16(
                false, a, false, bf, (short)0, acc[j], false, false);
        }
    }
    int mbase = mt * 16;
#pragma unroll
    for (int j = 0; j < 4; j++) {
        int col = ng * 64 + j * 16 + row;
        float bv = bias[col];
#pragma unroll
        for (int r = 0; r < 8; r++) {
            int m = mbase + r + 8 * half;
            gx[(size_t)m * G3H + col] = acc[j][r] + bv;
        }
    }
}

// ---------------- GH = weighted @ w_hh^T + b_hh  (per step, WMMA) ----------------
__global__ void gemm_gh_kernel(const unsigned short* __restrict__ wtd_bf,
                               const unsigned short* __restrict__ pW,
                               const float* __restrict__ bias,
                               float* __restrict__ gh) {
    int lane = threadIdx.x & 31;
    int wid  = threadIdx.x >> 5;
    int wg   = blockIdx.x * 8 + wid;          // 32 Mtiles * 12 Ngroups = 384
    int ng   = wg % 12;
    int mt   = wg / 12;
    int half = lane >> 4, row = lane & 15;
    const unsigned short* ar = wtd_bf + (size_t)(mt * 16 + row) * H_;
    v8f acc[4] = {};
#pragma unroll
    for (int kc = 0; kc < 8; ++kc) {
        v16bf a = load_a_bf(ar, kc * 32, half);
#pragma unroll
        for (int j = 0; j < 4; j++) {
            v16bf bf = load_b_frag(pW, 256, ng * 4 + j, kc, lane);
            acc[j] = __builtin_amdgcn_wmma_f32_16x16x32_bf16(
                false, a, false, bf, (short)0, acc[j], false, false);
        }
    }
    int mbase = mt * 16;
#pragma unroll
    for (int j = 0; j < 4; j++) {
        int col = ng * 64 + j * 16 + row;
        float bv = bias[col];
#pragma unroll
        for (int r = 0; r < 8; r++) {
            int m = mbase + r + 8 * half;
            gh[(size_t)m * G3H + col] = acc[j][r] + bv;
        }
    }
}

// ---------------- fused = [hall, static] @ fus_w^T + fus_b  (WMMA, K=320) -------
// A row m = t*B + b: k<256 from hall_bf (layout [T][B][H] => contiguous at m*H),
// k>=256 from static_bf
__global__ void gemm_fused_kernel(const unsigned short* __restrict__ hall_bf,
                                  const unsigned short* __restrict__ stat_bf,
                                  const unsigned short* __restrict__ pW,
                                  const float* __restrict__ bias,
                                  float* __restrict__ outf) {
    int lane = threadIdx.x & 31;
    int wid  = threadIdx.x >> 5;
    int wg   = blockIdx.x * 8 + wid;          // 4096 Mtiles * 4 Ngroups = 16384
    int ng   = wg & 3;
    int mt   = wg >> 2;
    int half = lane >> 4, row = lane & 15;
    int mrow = mt * 16 + row;
    int bb   = mrow & (B_ - 1);
    const unsigned short* arh = hall_bf + (size_t)mrow * H_;
    const unsigned short* ars = stat_bf + (size_t)bb * S_;
    v8f acc[4] = {};
#pragma unroll
    for (int kc = 0; kc < 10; ++kc) {
        int k0 = kc * 32;
        v16bf a = (k0 < 256) ? load_a_bf(arh, k0, half)
                             : load_a_bf(ars, k0 - 256, half);
#pragma unroll
        for (int j = 0; j < 4; j++) {
            v16bf bf = load_b_frag(pW, 320, ng * 4 + j, kc, lane);
            acc[j] = __builtin_amdgcn_wmma_f32_16x16x32_bf16(
                false, a, false, bf, (short)0, acc[j], false, false);
        }
    }
    int mbase = mt * 16;
#pragma unroll
    for (int j = 0; j < 4; j++) {
        int col = ng * 64 + j * 16 + row;
        float bv = bias[col];
#pragma unroll
        for (int r = 0; r < 8; r++) {
            int m = mbase + r + 8 * half;
            int b = m & (B_ - 1);
            int t = m >> 9;
            outf[((size_t)b * T_ + t) * H_ + col] = acc[j][r] + bv;
        }
    }
}

// ---------------- h0 = static @ init_h_w^T + init_h_b (into `weighted`) ---------
__global__ void h0_kernel(const float* __restrict__ stat,
                          const float* __restrict__ w,
                          const float* __restrict__ bias,
                          float* __restrict__ out,
                          unsigned short* __restrict__ out_bf) {
    int idx = blockIdx.x * blockDim.x + threadIdx.x;   // B*H
    if (idx >= B_ * H_) return;
    int b = idx >> 8, j = idx & 255;
    const float* s  = stat + (size_t)b * S_;
    const float* wr = w + (size_t)j * S_;
    float acc = bias[j];
#pragma unroll 8
    for (int i = 0; i < S_; i++) acc += s[i] * wr[i];
    out[idx] = acc;
    out_bf[idx] = f2bf_bits(acc);
}

// ---------------- per-step agent sampling + weighted hidden ---------------------
__global__ void sample_kernel(const float* __restrict__ x, const float* __restrict__ stat,
                              const float* __restrict__ hall,
                              const float* __restrict__ a1w1, const float* __restrict__ a1b1,
                              const float* __restrict__ a1w2, const float* __restrict__ a1b2,
                              const float* __restrict__ a2w1, const float* __restrict__ a2b1,
                              const float* __restrict__ a2w2, const float* __restrict__ a2b2,
                              float* __restrict__ weighted,
                              unsigned short* __restrict__ weighted_bf, int t) {
    __shared__ float obs1[H_ + S_], obs2[H_ + S_];
    __shared__ float hid1[U_], hid2[U_];
    __shared__ float lg1[K_], lg2[K_];
    __shared__ int   act[2];
    int b   = blockIdx.x;
    int tid = threadIdx.x;       // 128 threads

    for (int j = tid; j < H_; j += 128) {
        float s = 0.f;
#pragma unroll
        for (int w = 0; w < K_; ++w) {
            int hi = t - K_ + w;
            if (hi >= 0) s += hall[((size_t)hi * B_ + b) * H_ + j];
        }
        obs1[j] = s * (1.0f / K_);
        obs2[j] = x[((size_t)b * T_ + t) * D_ + j];
    }
    for (int j = tid; j < S_; j += 128) {
        float sv = stat[(size_t)b * S_ + j];
        obs1[H_ + j] = sv;
        obs2[H_ + j] = sv;
    }
    __syncthreads();

    if (tid < U_) {
        const float* wr = a1w1 + (size_t)tid * (H_ + S_);
        float acc = a1b1[tid];
        for (int i = 0; i < H_ + S_; i++) acc += obs1[i] * wr[i];
        hid1[tid] = tanhf(acc);
    } else {
        int u = tid - U_;
        const float* wr = a2w1 + (size_t)u * (D_ + S_);
        float acc = a2b1[u];
        for (int i = 0; i < D_ + S_; i++) acc += obs2[i] * wr[i];
        hid2[u] = tanhf(acc);
    }
    __syncthreads();

    if (tid < K_) {
        const float* wr = a1w2 + (size_t)tid * U_;
        float acc = a1b2[tid];
#pragma unroll 8
        for (int i = 0; i < U_; i++) acc += hid1[i] * wr[i];
        lg1[tid] = acc;
    } else if (tid >= 16 && tid < 16 + K_) {
        int k = tid - 16;
        const float* wr = a2w2 + (size_t)k * U_;
        float acc = a2b2[k];
#pragma unroll 8
        for (int i = 0; i < U_; i++) acc += hid2[i] * wr[i];
        lg2[k] = acc;
    }
    __syncthreads();

    if (tid < 2) {
        const float* lg = tid ? lg2 : lg1;
        unsigned c1 = 2u * (unsigned)t + (unsigned)tid;
        int best = 0;
        float bv = -1e30f;
        for (int k = 0; k < K_; k++) {
            unsigned r = threefry(42u, c1, (unsigned)b, (unsigned)k);
            float u01 = ((float)(r >> 8) + 0.5f) * (1.0f / 16777216.0f);
            float g = -__logf(-__logf(u01));
            float v = lg[k] + g;
            if (v > bv) { bv = v; best = k; }
        }
        act[tid] = best;
    }
    __syncthreads();

    int a1 = act[0], a2 = act[1];
    int i1 = t - K_ + a1, i2 = t - K_ + a2;
    const float* hprev = hall + ((size_t)(t - 1) * B_ + b) * H_;
    const float* h1p = (i1 >= 0) ? hall + ((size_t)i1 * B_ + b) * H_ : nullptr;
    const float* h2p = (i2 >= 0) ? hall + ((size_t)i2 * B_ + b) * H_ : nullptr;
    for (int j = tid; j < H_; j += 128) {
        float v1 = h1p ? h1p[j] : 0.f;
        float v2 = h2p ? h2p[j] : 0.f;
        float action_h = 0.5f * (v1 + v2);
        float wv = LAMBDA * action_h + (1.f - LAMBDA) * hprev[j];
        weighted[(size_t)b * H_ + j]    = wv;
        weighted_bf[(size_t)b * H_ + j] = f2bf_bits(wv);
    }
}

// ---------------- GRU gate fusion: h_t from gx[t], gh, weighted -----------------
__global__ void gate_kernel(const float* __restrict__ gx, const float* __restrict__ gh,
                            const float* __restrict__ weighted,
                            float* __restrict__ hall_t,
                            unsigned short* __restrict__ hall_bf_t) {
    int idx = blockIdx.x * blockDim.x + threadIdx.x;   // B*H
    if (idx >= B_ * H_) return;
    int b = idx >> 8, j = idx & 255;
    const float* gxr = gx + (size_t)b * G3H;
    const float* ghr = gh + (size_t)b * G3H;
    float r = 1.f / (1.f + __expf(-(gxr[j] + ghr[j])));
    float z = 1.f / (1.f + __expf(-(gxr[H_ + j] + ghr[H_ + j])));
    float n = tanhf(gxr[2 * H_ + j] + r * ghr[2 * H_ + j]);
    float h = (1.f - z) * n + z * weighted[idx];
    hall_t[idx]    = h;
    hall_bf_t[idx] = f2bf_bits(h);
}

// ---------------- last = fused[b, clip(sum(mask)-1)] ----------------------------
__global__ void last_kernel(const int* __restrict__ mask, const float* __restrict__ fused,
                            float* __restrict__ last) {
    __shared__ int sidx;
    int b = blockIdx.x;
    if (threadIdx.x == 0) {
        int s = 0;
        for (int t = 0; t < T_; t++) s += mask[(size_t)b * T_ + t];
        int idx = s - 1;
        if (idx < 0) idx = 0;
        if (idx > T_ - 1) idx = T_ - 1;
        sidx = idx;
    }
    __syncthreads();
    int idx = sidx;
    for (int j = threadIdx.x; j < H_; j += blockDim.x)
        last[(size_t)b * H_ + j] = fused[((size_t)b * T_ + idx) * H_ + j];
}

// ---------------- host launcher --------------------------------------------------
extern "C" void kernel_launch(void* const* d_in, const int* in_sizes, int n_in,
                              void* d_out, int out_size, void* d_ws, size_t ws_size,
                              hipStream_t stream) {
    (void)in_sizes; (void)n_in; (void)out_size; (void)ws_size;

    const float* x       = (const float*)d_in[0];
    const float* stat    = (const float*)d_in[1];
    const int*   mask    = (const int*)d_in[2];
    const float* a1w1    = (const float*)d_in[3];
    const float* a1b1    = (const float*)d_in[4];
    const float* a1w2    = (const float*)d_in[5];
    const float* a1b2    = (const float*)d_in[6];
    const float* a2w1    = (const float*)d_in[7];
    const float* a2b1    = (const float*)d_in[8];
    const float* a2w2    = (const float*)d_in[9];
    const float* a2b2    = (const float*)d_in[10];
    const float* w_ih    = (const float*)d_in[11];
    const float* w_hh    = (const float*)d_in[12];
    const float* b_ih    = (const float*)d_in[13];
    const float* b_hh    = (const float*)d_in[14];
    const float* init_hw = (const float*)d_in[15];
    const float* init_hb = (const float*)d_in[16];
    const float* fus_w   = (const float*)d_in[17];
    const float* fus_b   = (const float*)d_in[18];

    float* out_last  = (float*)d_out;
    float* out_fused = out_last + (size_t)B_ * H_;

    char* wsb = (char*)d_ws;
    size_t off = 0;
    auto alloc = [&](size_t bytes) -> char* {
        char* p = wsb + off;
        off += (bytes + 255) & ~(size_t)255;
        return p;
    };
    float*          GX      = (float*)alloc((size_t)T_ * B_ * G3H * sizeof(float));   // 192 MB
    float*          hall    = (float*)alloc((size_t)T_ * B_ * H_ * sizeof(float));    //  64 MB
    unsigned short* hall_bf = (unsigned short*)alloc((size_t)T_ * B_ * H_ * 2);       //  32 MB
    unsigned short* xbf     = (unsigned short*)alloc((size_t)B_ * T_ * D_ * 2);       //  32 MB
    float*          GH      = (float*)alloc((size_t)B_ * G3H * sizeof(float));
    float*          wtd     = (float*)alloc((size_t)B_ * H_ * sizeof(float));
    unsigned short* wtd_bf  = (unsigned short*)alloc((size_t)B_ * H_ * 2);
    unsigned short* stat_bf = (unsigned short*)alloc((size_t)B_ * S_ * 2);
    unsigned short* pWih    = (unsigned short*)alloc((size_t)G3H * D_ * 2);
    unsigned short* pWhh    = (unsigned short*)alloc((size_t)G3H * H_ * 2);
    unsigned short* pFus    = (unsigned short*)alloc((size_t)H_ * (H_ + S_) * 2);

    // 1) pack weights into WMMA B-fragment layout; build bf16 mirrors of A operands
    pack_b_kernel<<<(G3H * D_ + 255) / 256, 256, 0, stream>>>(w_ih, pWih, G3H, D_);
    pack_b_kernel<<<(G3H * H_ + 255) / 256, 256, 0, stream>>>(w_hh, pWhh, G3H, H_);
    pack_b_kernel<<<(H_ * (H_ + S_) + 255) / 256, 256, 0, stream>>>(fus_w, pFus, H_, H_ + S_);
    cvt_bf_kernel<<<(B_ * T_ * D_ + 255) / 256, 256, 0, stream>>>(x, xbf, B_ * T_ * D_);
    cvt_bf_kernel<<<(B_ * S_ + 255) / 256, 256, 0, stream>>>(stat, stat_bf, B_ * S_);

    // 2) GX for all timesteps (big WMMA GEMM, fully parallel)
    gemm_gx_kernel<<<(4096 * 12) / 8, 256, 0, stream>>>(xbf, pWih, b_ih, GX);

    // 3) h0 -> weighted (+ bf16 mirror)
    h0_kernel<<<(B_ * H_ + 255) / 256, 256, 0, stream>>>(stat, init_hw, init_hb, wtd, wtd_bf);

    // 4) recurrence: t = 0 uses weighted = h0; t >= 1 samples actions first
    for (int t = 0; t < T_; ++t) {
        if (t > 0) {
            sample_kernel<<<B_, 128, 0, stream>>>(x, stat, hall,
                                                  a1w1, a1b1, a1w2, a1b2,
                                                  a2w1, a2b1, a2w2, a2b2,
                                                  wtd, wtd_bf, t);
        }
        gemm_gh_kernel<<<(32 * 12) / 8, 256, 0, stream>>>(wtd_bf, pWhh, b_hh, GH);
        gate_kernel<<<(B_ * H_ + 255) / 256, 256, 0, stream>>>(
            GX + (size_t)t * B_ * G3H, GH, wtd,
            hall + (size_t)t * B_ * H_, hall_bf + (size_t)t * B_ * H_);
    }

    // 5) fused projection (big WMMA GEMM)
    gemm_fused_kernel<<<(4096 * 4) / 8, 256, 0, stream>>>(hall_bf, stat_bf, pFus, fus_b, out_fused);

    // 6) last-timestep gather
    last_kernel<<<B_, 128, 0, stream>>>(mask, out_fused, out_last);
}
